// Dual_Enconder_7584912244774
// MI455X (gfx1250) — matrix operand order repaired and checked by reading, not compiled, not run on hardware
//
#include <hip/hip_runtime.h>
#include <hip/hip_bf16.h>

// ---------------------------------------------------------------------------
// MI455X (gfx1250) implementation of the dual-encoder reference.
//
// Roofline: ~40 GFLOP of 128x128 GEMMs over 49152 tokens vs ~0.5 GB of HBM
// traffic (23.3 TB/s -> ~20us). Compute-bound on the linears -> route them
// through v_wmma_f32_16x16x32_bf16 (f32 accumulate, bf16 operands keep f32's
// exponent range). One wave32 owns a 16x16 f32 accumulator (8 VGPRs); a
// 256-thread block (8 waves) produces a 128x128 output tile.
//
// v2: the inner loops no longer do software bf16 rounding. Weights are
// converted once (hardware v_cvt_pk_bf16_f32) into bf16 LDS tiles, so each
// B fragment is 16 strided 16-bit LDS loads; A fragments use packed hw converts.
// ---------------------------------------------------------------------------

#define H_    8
#define DH    16
#define FEAT  128
#define B_    8
#define T_    12
#define N_    512
#define J_    16
#define SAMP  18
#define KSZ   3
#define MTOK  (B_ * T_ * N_)          // 49152 tokens
#define BETA_ 0.3f

typedef __attribute__((ext_vector_type(16))) __bf16         v16bf;
typedef __attribute__((ext_vector_type(16))) unsigned short v16u;
typedef __attribute__((ext_vector_type(8)))  unsigned int   v8u;
typedef __attribute__((ext_vector_type(8)))  float          v8f;

union BF16Frag { v16bf h; v16u u; v8u w; };

// two f32 -> packed bf16 pair (lowers to v_cvt_pk_bf16_f32)
__device__ __forceinline__ unsigned int pk_bf16(float a, float b) {
  union { __bf16 h[2]; unsigned int u; } p;
  p.h[0] = (__bf16)a;
  p.h[1] = (__bf16)b;
  return p.u;
}

__device__ __forceinline__ v8f v8f_zero() {
  v8f z;
#pragma unroll
  for (int i = 0; i < 8; ++i) z[i] = 0.0f;
  return z;
}

// ---------------------------------------------------------------------------
// Generic fused GEMM: C[M x 128] = A[M x 128] @ W[128 x 128] + bias
// (+ optional residual, optional ReLU).  W is row-major (fin, fout).
// Block: 256 threads = 8 waves; block tile = 128 rows x 128 cols.
// W converted to bf16 in LDS once; B frags = 16 strided 16-bit LDS loads.
// ---------------------------------------------------------------------------
__global__ __launch_bounds__(256) void k_gemm128(
    const float* __restrict__ A,
    const float* __restrict__ W,
    const float* __restrict__ bias,
    const float* __restrict__ res,
    float* __restrict__ C,
    int relu)
{
  __shared__ __attribute__((aligned(32))) unsigned short sW[FEAT * FEAT]; // 32 KB bf16
  const int tid = threadIdx.x;
  for (int e = tid * 4; e < FEAT * FEAT; e += 256 * 4) {
    const float4 w4 = *reinterpret_cast<const float4*>(&W[e]);
    uint2 p;
    p.x = pk_bf16(w4.x, w4.y);
    p.y = pk_bf16(w4.z, w4.w);
    *reinterpret_cast<uint2*>(&sW[e]) = p;
  }
  __syncthreads();

  const int wid = tid >> 5, lane = tid & 31;
  const int hl = lane >> 4, ml = lane & 15;
  const int row0 = blockIdx.x * 128 + wid * 16;

  v8f acc[8];
#pragma unroll
  for (int i = 0; i < 8; ++i) acc[i] = v8f_zero();

  const float* arow = A + (size_t)(row0 + ml) * FEAT;
  __builtin_prefetch(arow, 0, 1);                         // global_prefetch_b8

#pragma unroll
  for (int kc = 0; kc < 4; ++kc) {
    const int kb = kc * 32;
    // A fragment: 16x32 bf16, ISA layout (lane = M row; interleaved K pairs)
    BF16Frag af;
#pragma unroll
    for (int v = 0; v < 8; ++v) {
      const int k = kb + ((v < 4) ? (hl * 8 + 2 * v)
                                  : (16 + hl * 8 + 2 * (v - 4)));
      const float2 f2 = *reinterpret_cast<const float2*>(&arow[k]);
      af.w[v] = pk_bf16(f2.x, f2.y);
    }
    const unsigned short* wrow = &sW[kb * FEAT + ml];
#pragma unroll
    for (int nt = 0; nt < 8; ++nt) {
      BF16Frag bfr;
#pragma unroll
      for (int s = 0; s < 16; ++s) {
        const int k = (s < 8) ? (hl * 8 + s) : (16 + hl * 8 + (s - 8));
        bfr.u[s] = wrow[k * FEAT + nt * 16];
      }
      acc[nt] = __builtin_amdgcn_wmma_f32_16x16x32_bf16(
          false, af.h, false, bfr.h, (short)0, acc[nt], false, false);
    }
  }

  // C/D layout: col = nt*16 + (lane&15); row = row0 + r + 8*(lane>>4)
#pragma unroll
  for (int nt = 0; nt < 8; ++nt) {
    const int col = nt * 16 + ml;
    const float bv = bias ? bias[col] : 0.0f;
#pragma unroll
    for (int r = 0; r < 8; ++r) {
      const int row = row0 + r + 8 * hl;
      float y = acc[nt][r] + bv;
      if (res) y += res[(size_t)row * FEAT + col];
      if (relu) y = fmaxf(y, 0.0f);
      C[(size_t)row * FEAT + col] = y;
    }
  }
}

// ---------------------------------------------------------------------------
// Gated causal temporal conv (KSZ=3) as 3 shifted WMMA GEMMs, two accumulator
// sets (filter/gate), epilogue: relu(tanh(f)*sigmoid(g) + x).
// Weights are (cout, cin, k) row-major; per-tap slices are re-staged into
// bf16 LDS tiles (K-major) so the inner loop is pure ds_load + wmma.
// ---------------------------------------------------------------------------
__global__ __launch_bounds__(256) void k_tcn(
    const float* __restrict__ X,
    const float* __restrict__ WF, const float* __restrict__ BiasF,
    const float* __restrict__ WG, const float* __restrict__ BiasG,
    float* __restrict__ Y)
{
  __shared__ __attribute__((aligned(32))) unsigned short sF[FEAT * FEAT]; // 32 KB
  __shared__ __attribute__((aligned(32))) unsigned short sG[FEAT * FEAT]; // 32 KB

  const int tid = threadIdx.x;
  const int wid = tid >> 5, lane = tid & 31;
  const int hl = lane >> 4, ml = lane & 15;
  const int row0 = blockIdx.x * 128 + wid * 16;
  const int row = row0 + ml;
  const int trow = (row / N_) % T_;

  v8f accF[8], accG[8];
#pragma unroll
  for (int i = 0; i < 8; ++i) { accF[i] = v8f_zero(); accG[i] = v8f_zero(); }

  for (int dk = 0; dk < KSZ; ++dk) {
    // stage tap-dk weight slices: sF[cin*128 + cout] = bf16(WF[cout, cin, dk])
    __syncthreads();                                      // protect prev reads
    for (int e = tid; e < FEAT * FEAT / 2; e += 256) {
      const int cin = e >> 6;
      const int cp = (e & 63) * 2;                        // cout pair
      const size_t o = (size_t)cp * (FEAT * KSZ) + (size_t)cin * KSZ + dk;
      *reinterpret_cast<unsigned int*>(&sF[cin * FEAT + cp]) =
          pk_bf16(WF[o], WF[o + FEAT * KSZ]);
      *reinterpret_cast<unsigned int*>(&sG[cin * FEAT + cp]) =
          pk_bf16(WG[o], WG[o + FEAT * KSZ]);
    }
    __syncthreads();

    const int shift = (KSZ - 1) - dk;
    const bool valid = (trow - shift) >= 0;
    const float* arow = X + ((long)row - (long)shift * N_) * FEAT;
#pragma unroll
    for (int kc = 0; kc < 4; ++kc) {
      const int kb = kc * 32;
      BF16Frag af;
#pragma unroll
      for (int v = 0; v < 8; ++v) {
        const int k = kb + ((v < 4) ? (hl * 8 + 2 * v)
                                    : (16 + hl * 8 + 2 * (v - 4)));
        float x0 = 0.0f, x1 = 0.0f;
        if (valid) {
          const float2 f2 = *reinterpret_cast<const float2*>(&arow[k]);
          x0 = f2.x; x1 = f2.y;
        }
        af.w[v] = pk_bf16(x0, x1);
      }
      const unsigned short* fr = &sF[kb * FEAT + ml];
      const unsigned short* gr = &sG[kb * FEAT + ml];
#pragma unroll
      for (int nt = 0; nt < 8; ++nt) {
        BF16Frag bF, bG;
#pragma unroll
        for (int s = 0; s < 16; ++s) {
          const int k = (s < 8) ? (hl * 8 + s) : (16 + hl * 8 + (s - 8));
          bF.u[s] = fr[k * FEAT + nt * 16];
          bG.u[s] = gr[k * FEAT + nt * 16];
        }
        accF[nt] = __builtin_amdgcn_wmma_f32_16x16x32_bf16(
            false, af.h, false, bF.h, (short)0, accF[nt], false, false);
        accG[nt] = __builtin_amdgcn_wmma_f32_16x16x32_bf16(
            false, af.h, false, bG.h, (short)0, accG[nt], false, false);
      }
    }
  }

#pragma unroll
  for (int nt = 0; nt < 8; ++nt) {
    const int col = nt * 16 + ml;
    const float bfv = BiasF[col], bgv = BiasG[col];
#pragma unroll
    for (int r = 0; r < 8; ++r) {
      const int rr = row0 + r + 8 * hl;
      const float xv = X[(size_t)rr * FEAT + col];
      const float fv = tanhf(accF[nt][r] + bfv);
      float gv = accG[nt][r] + bgv;
      gv = 1.0f / (1.0f + __expf(-gv));
      Y[(size_t)rr * FEAT + col] = fmaxf(fv * gv + xv, 0.0f);
    }
  }
}

// ---------------------------------------------------------------------------
// Temporal attention core: per (h,b,n) a T=12 causal softmax attention with
// per-head focus factor. One wave per (h,b,n); K/V tiles staged in LDS.
// ---------------------------------------------------------------------------
__global__ __launch_bounds__(256) void k_temporal_attn(
    const float* __restrict__ Q, const float* __restrict__ K,
    const float* __restrict__ V, float* __restrict__ O)
{
  __shared__ float sK[8][T_ * DH];
  __shared__ float sV[8][T_ * DH];
  const int tid = threadIdx.x, wid = tid >> 5, lane = tid & 31;
  const int g = blockIdx.x * 8 + wid;                     // over H*B*N
  if (g >= H_ * B_ * N_) return;
  const int h = g / (B_ * N_);
  const int rem = g % (B_ * N_);
  const int bb = rem / N_;
  const int n = rem % N_;
  const size_t tstride = (size_t)N_ * FEAT;
  const size_t base = ((size_t)(bb * T_) * N_ + n) * FEAT + h * DH;

  for (int e = lane; e < T_ * DH; e += 32) {
    const int tt = e >> 4, d = e & 15;
    sK[wid][e] = K[base + tt * tstride + d];
    sV[wid][e] = V[base + tt * tstride + d];
  }
  __syncthreads();

  if (lane < T_) {
    const int t = lane;
    float qr[DH];
#pragma unroll
    for (int d = 0; d < DH; ++d) qr[d] = Q[base + t * tstride + d];
    const int fs = h, fe = (h == H_ - 1) ? T_ : h + 1;    // seg = T/H = 1
    float sc[T_];
    float mx = -INFINITY;
    for (int s = 0; s <= t; ++s) {
      float dot = 0.0f;
#pragma unroll
      for (int d = 0; d < DH; ++d) dot += qr[d] * sK[wid][s * DH + d];
      dot *= 0.25f;                                       // 1/sqrt(16)
      const float fac = (s >= fs && s < fe) ? 1.0f : BETA_;
      sc[s] = dot * fac;
      mx = fmaxf(mx, sc[s]);
    }
    float sum = 0.0f;
    for (int s = 0; s <= t; ++s) { sc[s] = __expf(sc[s] - mx); sum += sc[s]; }
    const float inv = 1.0f / sum;
    float o[DH];
#pragma unroll
    for (int d = 0; d < DH; ++d) o[d] = 0.0f;
    for (int s = 0; s <= t; ++s) {
      const float a = sc[s] * inv;
#pragma unroll
      for (int d = 0; d < DH; ++d) o[d] += a * sV[wid][s * DH + d];
    }
#pragma unroll
    for (int d = 0; d < DH; ++d) O[base + t * tstride + d] = o[d];
  }
}

// ---------------------------------------------------------------------------
// LayerNorm over the 128-feature axis (one wave per token), optional residual
// input and optional affine (gamma/beta).
// ---------------------------------------------------------------------------
__global__ __launch_bounds__(256) void k_ln(
    const float* __restrict__ X, const float* __restrict__ Res,
    const float* __restrict__ G, const float* __restrict__ Bb,
    float* __restrict__ Y, int M)
{
  const int wid = threadIdx.x >> 5, lane = threadIdx.x & 31;
  const int tok = blockIdx.x * 8 + wid;
  if (tok >= M) return;
  const float* x = X + (size_t)tok * FEAT;
  const float* r = Res ? Res + (size_t)tok * FEAT : nullptr;
  float v[4], s = 0.0f, sq = 0.0f;
#pragma unroll
  for (int j = 0; j < 4; ++j) {
    const int c = lane * 4 + j;
    v[j] = x[c] + (r ? r[c] : 0.0f);
    s += v[j]; sq += v[j] * v[j];
  }
  for (int o = 16; o >= 1; o >>= 1) {
    s += __shfl_xor(s, o);
    sq += __shfl_xor(sq, o);
  }
  const float mu = s * (1.0f / FEAT);
  const float var = sq * (1.0f / FEAT) - mu * mu;
  const float inv = rsqrtf(var + 1e-5f);
#pragma unroll
  for (int j = 0; j < 4; ++j) {
    const int c = lane * 4 + j;
    float y = (v[j] - mu) * inv;
    if (G) y = y * G[c] + Bb[c];
    Y[(size_t)tok * FEAT + c] = y;
  }
}

// ---------------------------------------------------------------------------
// Sparse spatial attention core: one block per (hb, t).
// sampling score -> top-18 -> masked softmax over N -> argmax gather.
// K tile in LDS; scores processed in 2 passes of 9 rows (<=64KB static LDS).
// ---------------------------------------------------------------------------
__global__ __launch_bounds__(256) void k_sparse_attn(
    const float* __restrict__ Q, const float* __restrict__ K,
    const float* __restrict__ V,
    const int*   __restrict__ LA,     // (N, J)
    const float* __restrict__ M1,     // (N, N)
    const float* __restrict__ M2,     // (H*N, N)
    const float* __restrict__ PW,     // (J,)
    const float* __restrict__ PB,     // (1,)
    float* __restrict__ O)
{
  __shared__ float sK[N_ * DH];        // 32 KB
  __shared__ float sM[N_];             //  2 KB
  __shared__ float sScore[9 * N_];     // 18 KB
  __shared__ float sQr[9 * DH];
  __shared__ float sRedV[256];
  __shared__ int   sRedI[256];
  __shared__ int   sTop[SAMP];
  __shared__ float sVal[SAMP * DH];
  __shared__ float sRunMax[N_];
  __shared__ int   sRunArg[N_];

  const int tid = threadIdx.x;
  const int hb = blockIdx.x / T_;
  const int t = blockIdx.x % T_;
  const int h = hb / B_, bb = hb % B_;
  const size_t base = ((size_t)(bb * T_ + t) * N_) * FEAT + h * DH;

  for (int e = tid; e < N_ * DH; e += 256) {
    const int n = e >> 4, d = e & 15;
    sK[e] = K[base + (size_t)n * FEAT + d];
  }
  __syncthreads();

  // sampling score m[n] = (sum_j (q[n].k[la[n,j]]) * pw[j]) + pb
  for (int n = tid; n < N_; n += 256) {
    float qr[DH];
#pragma unroll
    for (int d = 0; d < DH; ++d) qr[d] = Q[base + (size_t)n * FEAT + d];
    float mv = PB[0];
    for (int j = 0; j < J_; ++j) {
      const int kj = LA[n * J_ + j];
      float dot = 0.0f;
#pragma unroll
      for (int d = 0; d < DH; ++d) dot += qr[d] * sK[kj * DH + d];
      mv += dot * PW[j];
    }
    sM[n] = mv;
    sRunMax[n] = -INFINITY;
    sRunArg[n] = 0;
  }
  __syncthreads();

  // top-18 by iterative block-wide argmax (ties -> smaller index, like top_k)
  for (int it = 0; it < SAMP; ++it) {
    float bv = -INFINITY; int bi = 0;
    for (int n = tid; n < N_; n += 256) {
      const float vv = sM[n];
      if (vv > bv) { bv = vv; bi = n; }
    }
    sRedV[tid] = bv; sRedI[tid] = bi;
    __syncthreads();
    for (int off = 128; off > 0; off >>= 1) {
      if (tid < off) {
        const float ov = sRedV[tid + off]; const int oi = sRedI[tid + off];
        if (ov > sRedV[tid] || (ov == sRedV[tid] && oi < sRedI[tid])) {
          sRedV[tid] = ov; sRedI[tid] = oi;
        }
      }
      __syncthreads();
    }
    if (tid == 0) { sTop[it] = sRedI[0]; sM[sRedI[0]] = -INFINITY; }
    __syncthreads();
  }

  // two passes of 9 sampled rows
  for (int p = 0; p < 2; ++p) {
    const int s0 = p * 9;
    if (tid < 9 * DH) {
      const int s = tid >> 4, d = tid & 15;
      sQr[tid] = Q[base + (size_t)sTop[s0 + s] * FEAT + d];
    }
    __syncthreads();

    for (int e = tid; e < 9 * N_; e += 256) {
      const int s = e / N_, n = e - s * N_;
      const int ms = sTop[s0 + s];
      float dot = 0.0f;
#pragma unroll
      for (int d = 0; d < DH; ++d) dot += sQr[s * DH + d] * sK[n * DH + d];
      float sc = dot * 0.25f;
      sc *= M1[(size_t)ms * N_ + n];
      sc *= M2[((size_t)h * N_ + ms) * (size_t)N_ + n];
      sScore[e] = sc;
    }
    __syncthreads();

    // softmax over N per row (wave-parallel)
    const int wid = tid >> 5, lane = tid & 31;
    for (int s = wid; s < 9; s += 8) {
      float* rowp = sScore + s * N_;
      float mx = -INFINITY;
      for (int n = lane; n < N_; n += 32) mx = fmaxf(mx, rowp[n]);
      for (int o = 16; o >= 1; o >>= 1) mx = fmaxf(mx, __shfl_xor(mx, o));
      float sum = 0.0f;
      for (int n = lane; n < N_; n += 32) {
        const float ev = __expf(rowp[n] - mx);
        rowp[n] = ev; sum += ev;
      }
      for (int o = 16; o >= 1; o >>= 1) sum += __shfl_xor(sum, o);
      const float inv = 1.0f / sum;
      for (int n = lane; n < N_; n += 32) rowp[n] *= inv;
    }
    __syncthreads();

    // running argmax over sampled axis per column n (first max wins, argmax)
    for (int n = tid; n < N_; n += 256) {
      float bm = sRunMax[n]; int ba = sRunArg[n];
      for (int s = 0; s < 9; ++s) {
        const float vv = sScore[s * N_ + n];
        if (vv > bm) { bm = vv; ba = s0 + s; }
      }
      sRunMax[n] = bm; sRunArg[n] = ba;
    }

    // val[s][d] = attn[s] . V[:, d]
    if (tid < 9 * DH) {
      const int s = tid >> 4, d = tid & 15;
      const float* rowp = sScore + s * N_;
      float acc = 0.0f;
      for (int n = 0; n < N_; ++n) acc += rowp[n] * V[base + (size_t)n * FEAT + d];
      sVal[(s0 + s) * DH + d] = acc;
    }
    __syncthreads();
  }

  // out[n] = val[cp[n]]
  for (int n = tid; n < N_; n += 256) {
    const int cp = sRunArg[n];
#pragma unroll
    for (int d = 0; d < DH; ++d)
      O[base + (size_t)n * FEAT + d] = sVal[cp * DH + d];
  }
}

// ---------------------------------------------------------------------------
// Elementwise helpers
// ---------------------------------------------------------------------------
__global__ void k_add4(const float4* __restrict__ a, const float4* __restrict__ b,
                       float4* __restrict__ c, int n4)
{
  const int i = blockIdx.x * blockDim.x + threadIdx.x;
  if (i >= n4) return;
  const float4 x = a[i], y = b[i];
  float4 z; z.x = x.x + y.x; z.y = x.y + y.y; z.z = x.z + y.z; z.w = x.w + y.w;
  c[i] = z;
}

__global__ void k_spa_pre(const float* __restrict__ svec, const float* __restrict__ sval,
                          const float* __restrict__ tvec, const float* __restrict__ tval,
                          float* __restrict__ addv)
{
  const int i = blockIdx.x * blockDim.x + threadIdx.x;
  if (i >= N_ * FEAT) return;
  const int f = i & (FEAT - 1);
  addv[i] = svec[i] * sval[f] + tvec[i] * tval[f];
}

__global__ void k_add_bcast(const float* __restrict__ x, const float* __restrict__ addv,
                            float* __restrict__ y, long total)
{
  const long i = (long)blockIdx.x * blockDim.x + threadIdx.x;
  if (i >= total) return;
  const long tok = i >> 7;
  const int f = (int)(i & (FEAT - 1));
  const int n = (int)(tok % N_);
  y[i] = x[i] + addv[n * FEAT + f];
}

// ---------------------------------------------------------------------------
// Host orchestration.
// Input order (depth-first insertion order of setup_inputs(), 'w' before 'b'):
//  0 xl, 1 xh, 2 te, 3 la(int), 4 mask1, 5 mask2_big,
//  6 spa_eigvalue, 7 spa_eigvec, 8 tem_eigvalue, 9 tem_eigvec,
//  ta: 10 qw,11 qb,12 kw,13 kb,14 vw,15 vb,16 ow,17 ob,18 f1w,19 f1b,20 f2w,21 f2b,
//  tcn: 22 wf,23 bf,24 wg,25 bg,
//  sa_l: 26 qw,27 qb,28 kw,29 kb,30 vw,31 vb,32 ow,33 ob,34 ffw,35 ffb,
//        36 projw,37 projb,38 ln_g,39 ln_b,
//  sa_h: 40..53 (same layout).
// ---------------------------------------------------------------------------
extern "C" void kernel_launch(void* const* d_in, const int* in_sizes, int n_in,
                              void* d_out, int out_size, void* d_ws, size_t ws_size,
                              hipStream_t stream) {
  (void)in_sizes; (void)n_in; (void)out_size; (void)ws_size;

  const size_t SZ = (size_t)MTOK * FEAT;      // 6,291,456 floats per buffer
  float* ws = (float*)d_ws;
  float* b0 = ws + 0 * SZ;                    // xpe / x_ (spa input)
  float* b1 = ws + 1 * SZ;                    // q
  float* b2 = ws + 2 * SZ;                    // k
  float* b3 = ws + 3 * SZ;                    // v
  float* b4 = ws + 4 * SZ;                    // attention output
  float* b5 = ws + 5 * SZ;                    // gemm scratch
  float* b6 = ws + 6 * SZ;                    // ln output
  float* b7 = ws + 7 * SZ;                    // ff hidden
  float* b8 = ws + 8 * SZ;                    // xl2
  float* b9 = ws + 9 * SZ;                    // xh2
  float* addspa = ws + 10 * SZ;               // (N, FEAT) eigen offset table

  const float* xl = (const float*)d_in[0];
  const float* xh = (const float*)d_in[1];
  const float* te = (const float*)d_in[2];
  const int*   la = (const int*)d_in[3];
  const float* m1 = (const float*)d_in[4];
  const float* m2 = (const float*)d_in[5];
  const float* sval = (const float*)d_in[6];
  const float* svec = (const float*)d_in[7];
  const float* tval = (const float*)d_in[8];
  const float* tvec = (const float*)d_in[9];
  auto P = [&](int i) { return (const float*)d_in[i]; };

  float* out = (float*)d_out;

  const int GEMM_BLKS = MTOK / 128;           // 384
  const int LN_BLKS   = MTOK / 8;             // 6144
  const int N4        = (int)(SZ / 4);

  auto gemm = [&](const float* A, const float* W, const float* bias,
                  const float* res, float* C, int relu) {
    k_gemm128<<<GEMM_BLKS, 256, 0, stream>>>(A, W, bias, res, C, relu);
  };

  // ---- Stage 1: temporal attention on xl -------------------------------
  k_add4<<<(N4 + 255) / 256, 256, 0, stream>>>(
      (const float4*)xl, (const float4*)te, (float4*)b0, N4);     // xpe
  gemm(b0, P(10), P(11), nullptr, b1, 0);                         // q
  gemm(b0, P(12), P(13), nullptr, b2, 0);                         // k
  gemm(b0, P(14), P(15), nullptr, b3, 0);                         // v
  k_temporal_attn<<<(H_ * B_ * N_) / 8, 256, 0, stream>>>(b1, b2, b3, b4);
  gemm(b4, P(16), P(17), nullptr, b5, 0);                         // ofc
  k_ln<<<LN_BLKS, 256, 0, stream>>>(b5, b0, nullptr, nullptr, b6, MTOK);
  gemm(b6, P(18), P(19), nullptr, b7, 1);                         // ff1 + relu
  gemm(b7, P(20), P(21), nullptr, b5, 0);                         // ff2
  k_ln<<<LN_BLKS, 256, 0, stream>>>(b5, b6, nullptr, nullptr, b8, MTOK);  // xl2

  // ---- Stage 2: gated temporal conv on xh ------------------------------
  k_tcn<<<GEMM_BLKS, 256, 0, stream>>>(xh, P(22), P(23), P(24), P(25), b9); // xh2

  // ---- Eigen-offset table (shared by both sparse blocks) ---------------
  k_spa_pre<<<(N_ * FEAT + 255) / 256, 256, 0, stream>>>(svec, sval, tvec, tval, addspa);

  // ---- Stage 3: sparse spatial attention (low + high) ------------------
  for (int s = 0; s < 2; ++s) {
    const int p0 = 26 + 14 * s;               // sa_l: 26.., sa_h: 40..
    const float* xin = (s == 0) ? b8 : b9;
    float* res_out = out + (size_t)s * SZ;

    k_add_bcast<<<(int)((SZ + 255) / 256), 256, 0, stream>>>(xin, addspa, b0, (long)SZ);
    gemm(b0, P(p0 + 0), P(p0 + 1), nullptr, b1, 0);               // q
    gemm(b0, P(p0 + 2), P(p0 + 3), nullptr, b2, 0);               // k
    gemm(b0, P(p0 + 4), P(p0 + 5), nullptr, b3, 0);               // v
    k_sparse_attn<<<H_ * B_ * T_, 256, 0, stream>>>(
        b1, b2, b3, la, m1, m2, P(p0 + 10), P(p0 + 11), b4);
    gemm(b4, P(p0 + 6), P(p0 + 7), nullptr, b5, 0);               // ofc
    k_ln<<<LN_BLKS, 256, 0, stream>>>(b5, nullptr, P(p0 + 12), P(p0 + 13), b6, MTOK);
    gemm(b6, P(p0 + 8), P(p0 + 9), xin, res_out, 0);              // ff + residual
  }
}
